// DMoNPoolingLayer_37761352466634
// MI455X (gfx1250) — matrix operand order repaired and checked by reading, not compiled, not run on hardware
//
#include <hip/hip_runtime.h>
#include <math.h>

typedef __bf16 bf16;
typedef __attribute__((ext_vector_type(16))) __bf16 v16bf;
typedef __attribute__((ext_vector_type(8)))  float  v8f;

#define N_NODES   131072
#define F_DIM_C   64
#define K_CL      16
#define B_GR      16
#define NPG       8192
#define SELU_SCALE 1.0507009873554805f
#define SELU_ALPHA 1.6732632423543772f

// ---------------- zero workspace region ----------------
__global__ void zero_kernel(float4* __restrict__ p, long long n16) {
  long long i = (long long)blockIdx.x * blockDim.x + threadIdx.x;
  if (i < n16) p[i] = make_float4(0.f, 0.f, 0.f, 0.f);
}

// ---------------- x [N,64] f32 -> xT [64,N] bf16 ----------------
__global__ void transpose_x_kernel(const float* __restrict__ x, bf16* __restrict__ xT) {
  int i = blockIdx.x * blockDim.x + threadIdx.x;  // node
  if (i >= N_NODES) return;
  const float* xr = x + (size_t)i * F_DIM_C;
#pragma unroll
  for (int f = 0; f < F_DIM_C; ++f)
    xT[(size_t)f * N_NODES + i] = (bf16)xr[f];    // coalesced across threads per f
}

// ---------------- s = x@W + b (WMMA), softmax -> ss/sT/ssT ----------------
// one wave per 16 nodes; block = 8 waves = 128 nodes
__global__ void assign_kernel(const float* __restrict__ x, const float* __restrict__ W,
                              const float* __restrict__ bias, float* __restrict__ ss,
                              bf16* __restrict__ sT, bf16* __restrict__ ssT) {
  __shared__ float tile[8][K_CL][K_CL + 1];
  const int wave = threadIdx.x >> 5;
  const int lane = threadIdx.x & 31;
  const int m    = lane & 15;
  const int hi   = lane >> 4;
  const int rowbase = blockIdx.x * 128 + wave * 16;

  // A fragments (16x32 bf16): lane holds row m, K runs {hi*8..+7} and {16+hi*8..+7}
  const float* xrow = x + (size_t)(rowbase + m) * F_DIM_C;
  v16bf a0, a1, b0, b1;
#pragma unroll
  for (int j = 0; j < 8; ++j) {
    a0[j]     = (bf16)xrow[hi * 8 + j];
    a0[j + 8] = (bf16)xrow[16 + hi * 8 + j];
    a1[j]     = (bf16)xrow[32 + hi * 8 + j];
    a1[j + 8] = (bf16)xrow[48 + hi * 8 + j];
  }
  // B fragments (32x16 bf16): lane holds column m, K runs {hi*8..+7} and {16+hi*8..+7}
#pragma unroll
  for (int j = 0; j < 8; ++j) {
    b0[j]     = (bf16)W[(hi * 8 + j) * K_CL + m];
    b0[j + 8] = (bf16)W[(16 + hi * 8 + j) * K_CL + m];
    b1[j]     = (bf16)W[(32 + hi * 8 + j) * K_CL + m];
    b1[j + 8] = (bf16)W[(48 + hi * 8 + j) * K_CL + m];
  }
  v8f c = {};
  c = __builtin_amdgcn_wmma_f32_16x16x32_bf16(false, a0, false, b0, (short)0, c, false, false);
  c = __builtin_amdgcn_wmma_f32_16x16x32_bf16(false, a1, false, b1, (short)0, c, false, false);

  const float bn = bias[m];
#pragma unroll
  for (int r = 0; r < 8; ++r) {               // C/D layout: row = r + 8*hi, col = m
    const int mm = r + 8 * hi;
    const float v = c[r] + bn;
    tile[wave][mm][m] = v;
    sT[(size_t)m * N_NODES + rowbase + mm] = (bf16)v;  // 8 consecutive nodes per lane
  }
  __syncthreads();
  if (hi == 0) {                              // lanes 0..15: softmax of row 'lane'
    const int mm = lane;
    float mx = -3.4e38f;
#pragma unroll
    for (int nn = 0; nn < K_CL; ++nn) mx = fmaxf(mx, tile[wave][mm][nn]);
    float e[K_CL]; float sum = 0.f;
#pragma unroll
    for (int nn = 0; nn < K_CL; ++nn) { e[nn] = __expf(tile[wave][mm][nn] - mx); sum += e[nn]; }
    const float inv = 1.f / sum;
    float* srow = ss + (size_t)(rowbase + mm) * K_CL;
#pragma unroll
    for (int nn = 0; nn < K_CL; ++nn) {
      const float p = e[nn] * inv;
      tile[wave][mm][nn] = p;
      srow[nn] = p;
    }
  }
  __syncthreads();
#pragma unroll
  for (int r = 0; r < 8; ++r) {
    const int mm = r + 8 * hi;
    ssT[(size_t)m * N_NODES + rowbase + mm] = (bf16)tile[wave][mm][m];
  }
}

// ---------------- d[col] += w ----------------
__global__ void degree_kernel(const float* __restrict__ w, const int* __restrict__ col,
                              float* __restrict__ d, int E) {
  int e = blockIdx.x * blockDim.x + threadIdx.x;
  if (e < E) atomicAdd(&d[col[e]], w[e]);
}

// ---------------- AC[row] += w * ss[col] ----------------
__global__ void scatter_kernel(const float* __restrict__ w, const int* __restrict__ row,
                               const int* __restrict__ col, const float* __restrict__ ss,
                               float* __restrict__ AC, int E) {
  int e = blockIdx.x * blockDim.x + threadIdx.x;
  if (e >= E) return;
  const float we = w[e];
  const float4* sp = (const float4*)(ss + (size_t)col[e] * K_CL);
  float* ap = AC + (size_t)row[e] * K_CL;
#pragma unroll
  for (int q = 0; q < 4; ++q) {
    float4 v = sp[q];
    atomicAdd(ap + 4 * q + 0, we * v.x);
    atomicAdd(ap + 4 * q + 1, we * v.y);
    atomicAdd(ap + 4 * q + 2, we * v.z);
    atomicAdd(ap + 4 * q + 3, we * v.w);
  }
}

// ---------------- AC [N,16] f32 -> ACT [16,N] bf16 ----------------
__global__ void act_kernel(const float* __restrict__ AC, bf16* __restrict__ ACT) {
  int i = blockIdx.x * blockDim.x + threadIdx.x;
  if (i >= N_NODES) return;
  const float* ar = AC + (size_t)i * K_CL;
#pragma unroll
  for (int k = 0; k < K_CL; ++k)
    ACT[(size_t)k * N_NODES + i] = (bf16)ar[k];
}

// ---------------- pooled einsums via WMMA, K=8192 reduction ----------------
// blockIdx = graph*6 + tile (0..3: out_x N-tiles, 4: out_adj, 5: CC); wave = K-chunk (1024 nodes)
__global__ void pool_wmma_kernel(const bf16* __restrict__ sT, const bf16* __restrict__ ssT,
                                 const bf16* __restrict__ xT, const bf16* __restrict__ ACT,
                                 float* __restrict__ outx, float* __restrict__ oadj,
                                 float* __restrict__ CC) {
  const int wave = threadIdx.x >> 5;
  const int lane = threadIdx.x & 31;
  const int m  = lane & 15;
  const int hi = lane >> 4;
  const int g = blockIdx.x / 6;
  const int t = blockIdx.x % 6;

  const bf16* Ap; const bf16* Bp; float* acc; int cols, colbase;
  if (t < 4)       { Ap = sT;  Bp = xT;  acc = outx + (size_t)g * K_CL * F_DIM_C; cols = F_DIM_C; colbase = t * 16; }
  else if (t == 4) { Ap = ssT; Bp = ACT; acc = oadj + (size_t)g * 256;            cols = K_CL;    colbase = 0; }
  else             { Ap = ssT; Bp = ssT; acc = CC   + (size_t)g * 256;            cols = K_CL;    colbase = 0; }

  const bf16* Arow = Ap + (size_t)m * N_NODES;                   // A row = cluster m
  const bf16* Brow = Bp + (size_t)(colbase + m) * N_NODES;       // B col = colbase + m
  const int kstart = g * NPG + wave * 1024;

  v8f c = {};
  for (int it = 0; it < 32; ++it) {
    const int kb = kstart + it * 32;
    v16bf a, b;
#pragma unroll
    for (int j = 0; j < 8; ++j) {
      a[j]     = Arow[kb + hi * 8 + j];
      a[j + 8] = Arow[kb + 16 + hi * 8 + j];
    }
#pragma unroll
    for (int j = 0; j < 8; ++j) {
      b[j]     = Brow[kb + hi * 8 + j];
      b[j + 8] = Brow[kb + 16 + hi * 8 + j];
    }
    c = __builtin_amdgcn_wmma_f32_16x16x32_bf16(false, a, false, b, (short)0, c, false, false);
  }
#pragma unroll
  for (int r = 0; r < 8; ++r)
    atomicAdd(&acc[(size_t)(r + 8 * hi) * cols + colbase + m], c[r]);
}

// ---------------- dS / cluster_size / m2 ----------------
__global__ void stats_kernel(const bf16* __restrict__ ssT, const float* __restrict__ d,
                             float* __restrict__ dS, float* __restrict__ csz,
                             float* __restrict__ m2) {
  __shared__ float r1[256], r2[256], r3[256];
  const int b = blockIdx.x >> 4;
  const int k = blockIdx.x & 15;
  const int tid = threadIdx.x;
  const bf16* sr = ssT + (size_t)k * N_NODES + (size_t)b * NPG;
  const float* dr = d + (size_t)b * NPG;
  float s1 = 0.f, s2 = 0.f, s3 = 0.f;
  for (int i = tid; i < NPG; i += 256) {
    const float sv = (float)sr[i];
    const float dv = dr[i];
    s1 += dv * sv; s2 += sv; s3 += dv;
  }
  r1[tid] = s1; r2[tid] = s2; r3[tid] = s3;
  __syncthreads();
  for (int st = 128; st > 0; st >>= 1) {
    if (tid < st) { r1[tid] += r1[tid + st]; r2[tid] += r2[tid + st]; r3[tid] += r3[tid + st]; }
    __syncthreads();
  }
  if (tid == 0) {
    dS[blockIdx.x]  = r1[0];
    csz[blockIdx.x] = r2[0];
    if (k == 0) m2[b] = r3[0];
  }
}

// ---------------- SELU, losses, out_adj_norm ----------------
__global__ void finalize_kernel(const float* __restrict__ outx, const float* __restrict__ oadj,
                                const float* __restrict__ CC, const float* __restrict__ dS,
                                const float* __restrict__ csz, const float* __restrict__ m2,
                                float* __restrict__ out) {
  __shared__ float spec[B_GR], clus[B_GR], orth[B_GR], dd[B_GR * K_CL];
  const int tid = threadIdx.x;
  for (int i = tid; i < B_GR * K_CL * F_DIM_C; i += 256) {
    const float v = outx[i];
    out[i] = v > 0.f ? SELU_SCALE * v : SELU_SCALE * SELU_ALPHA * (__expf(v) - 1.f);
  }
  if (tid < B_GR) {
    const int b = tid;
    const float mb = m2[b];
    float tr = 0.f, d2 = 0.f, cs2 = 0.f, ccss = 0.f;
    for (int k = 0; k < K_CL; ++k) {
      tr += oadj[b * 256 + k * K_CL + k];
      const float dv = dS[b * K_CL + k];  d2  += dv * dv;
      const float cv = csz[b * K_CL + k]; cs2 += cv * cv;
    }
    spec[b] = tr / mb - d2 / (mb * mb);
    clus[b] = sqrtf(cs2) / (float)NPG * 4.0f - 1.0f;   // sqrt(K)=4
    for (int q = 0; q < 256; ++q) { const float v = CC[b * 256 + q]; ccss += v * v; }
    const float ccn = sqrtf(ccss);
    float os = 0.f;
    for (int q = 0; q < 256; ++q) {
      const float tgt = ((q >> 4) == (q & 15)) ? 0.25f : 0.f;  // I/sqrt(K)
      const float v = CC[b * 256 + q] / ccn - tgt;
      os += v * v;
    }
    orth[b] = sqrtf(os);
  }
  for (int idx = tid; idx < B_GR * K_CL; idx += 256) {
    const int b = idx >> 4, k = idx & 15;
    float rs = 0.f;
    for (int l = 0; l < K_CL; ++l) if (l != k) rs += oadj[b * 256 + k * K_CL + l];
    dd[idx] = sqrtf(rs) + 1e-12f;
  }
  __syncthreads();
  for (int idx = tid; idx < B_GR * 256; idx += 256) {
    const int b = idx >> 8, k = (idx >> 4) & 15, l = idx & 15;
    const float oa = (k == l) ? 0.f : oadj[idx];
    out[B_GR * K_CL * F_DIM_C + idx] = oa / (dd[b * K_CL + k] * dd[b * K_CL + l]);
  }
  if (tid == 0) {
    float sp = 0.f, cl = 0.f, orr = 0.f;
    for (int b = 0; b < B_GR; ++b) { sp += spec[b]; cl += clus[b]; orr += orth[b]; }
    const int base = B_GR * K_CL * F_DIM_C + B_GR * 256;
    out[base + 0] = -sp / B_GR;
    out[base + 1] =  cl / B_GR;
    out[base + 2] = orr / B_GR;
  }
}

extern "C" void kernel_launch(void* const* d_in, const int* in_sizes, int n_in,
                              void* d_out, int out_size, void* d_ws, size_t ws_size,
                              hipStream_t stream) {
  const float* x  = (const float*)d_in[0];
  const float* W  = (const float*)d_in[1];
  const float* bb = (const float*)d_in[2];
  const float* ew = (const float*)d_in[3];
  const int*   er = (const int*)d_in[4];
  const int*   ec = (const int*)d_in[5];
  const int E = in_sizes[3];
  float* out = (float*)d_out;
  (void)n_in; (void)out_size; (void)ws_size;

  char* ws = (char*)d_ws;
  size_t off = 0;
  auto take = [&](size_t bytes) -> char* {
    char* p = ws + off;
    off = (off + bytes + 255) & ~(size_t)255;
    return p;
  };
  // zero-init region first
  float* AC   = (float*)take((size_t)N_NODES * K_CL * 4);
  float* dg   = (float*)take((size_t)N_NODES * 4);
  float* outx = (float*)take((size_t)B_GR * K_CL * F_DIM_C * 4);
  float* oadj = (float*)take((size_t)B_GR * 256 * 4);
  float* CCa  = (float*)take((size_t)B_GR * 256 * 4);
  float* dSa  = (float*)take((size_t)B_GR * K_CL * 4);
  float* csz  = (float*)take((size_t)B_GR * K_CL * 4);
  float* m2a  = (float*)take((size_t)B_GR * 4);
  const size_t zeroBytes = off;
  float* ssf  = (float*)take((size_t)N_NODES * K_CL * 4);
  bf16*  sT   = (bf16*)take((size_t)K_CL * N_NODES * 2);
  bf16*  ssT  = (bf16*)take((size_t)K_CL * N_NODES * 2);
  bf16*  ACT  = (bf16*)take((size_t)K_CL * N_NODES * 2);
  bf16*  xT   = (bf16*)take((size_t)F_DIM_C * N_NODES * 2);

  {
    long long n16 = (long long)(zeroBytes / 16);
    int blocks = (int)((n16 + 255) / 256);
    zero_kernel<<<blocks, 256, 0, stream>>>((float4*)ws, n16);
  }
  transpose_x_kernel<<<N_NODES / 256, 256, 0, stream>>>(x, xT);
  assign_kernel<<<N_NODES / 128, 256, 0, stream>>>(x, W, bb, ssf, sT, ssT);
  degree_kernel<<<(E + 255) / 256, 256, 0, stream>>>(ew, ec, dg, E);
  scatter_kernel<<<(E + 255) / 256, 256, 0, stream>>>(ew, er, ec, ssf, AC, E);
  act_kernel<<<N_NODES / 256, 256, 0, stream>>>(AC, ACT);
  pool_wmma_kernel<<<B_GR * 6, 256, 0, stream>>>(sT, ssT, xT, ACT, outx, oadj, CCa);
  stats_kernel<<<B_GR * K_CL, 256, 0, stream>>>(ssT, dg, dSa, csz, m2a);
  finalize_kernel<<<1, 256, 0, stream>>>(outx, oadj, CCa, dSa, csz, m2a, out);
}
